// TH_Attention_77094662963330
// MI455X (gfx1250) — matrix operand order repaired and checked by reading, not compiled, not run on hardware
//
#include <hip/hip_runtime.h>
#include <hip/hip_bf16.h>

#define B_ 32
#define T_ 8192
#define H_ 256
#define LOG2E_F 1.4426950408889634f
#define WPAD 264            // 256 + 8 bf16 elems; row stride 528B stays 16B-aligned
#define NUM_OFF (H_ * WPAD * 2)          // 135168: float numLDS[256]
#define DEN_OFF (NUM_OFF + H_ * 4)       // 136192: float denLDS
#define LDS_BYTES (DEN_OFF + 16)

typedef __bf16 v16bf __attribute__((ext_vector_type(16)));
typedef float  v8f   __attribute__((ext_vector_type(8)));

struct U32x8 { unsigned u[8]; };

#if __has_builtin(__builtin_amdgcn_global_load_async_to_lds_b128)
#define HAVE_ASYNC_LDS 1
typedef int v4i_raw __attribute__((vector_size(4 * sizeof(int))));
typedef __attribute__((address_space(1))) v4i_raw* g_v4i_ptr;
typedef __attribute__((address_space(3))) v4i_raw* l_v4i_ptr;
#endif

// round-to-nearest-even f32 -> bf16, two at a time packed into a dword
static __device__ __forceinline__ unsigned pack2bf(float a, float b) {
    unsigned ua = __float_as_uint(a);
    unsigned ub = __float_as_uint(b);
    ua += 0x7FFFu + ((ua >> 16) & 1u);
    ub += 0x7FFFu + ((ub >> 16) & 1u);
    return (ua >> 16) | (ub & 0xFFFF0000u);
}

static __device__ __forceinline__ unsigned short f32_to_bf16(float a) {
    unsigned u = __float_as_uint(a);
    u += 0x7FFFu + ((u >> 16) & 1u);
    return (unsigned short)(u >> 16);
}

static __device__ __forceinline__ float fast_tanh(float x) {
#if __has_builtin(__builtin_amdgcn_tanhf)
    return __builtin_amdgcn_tanhf(x);
#else
    float r;
    asm volatile("v_tanh_f32 %0, %1\n\tv_nop\n\tv_nop" : "=v"(r) : "v"(x));
    return r;
#endif
}

static __device__ __forceinline__ void wait_async_then_barrier() {
#if defined(HAVE_ASYNC_LDS)
#if __has_builtin(__builtin_amdgcn_s_wait_asynccnt)
    __builtin_amdgcn_s_wait_asynccnt(0);
#else
    asm volatile("s_wait_asynccnt 0x0" ::: "memory");
#endif
#endif
    __syncthreads();
}

// ---------------------------------------------------------------------------
// K0: zero the f32 accumulators (num[B*H], den[B]) — must run every launch.
__global__ void init_kernel(float* __restrict__ num, float* __restrict__ den) {
    int i = blockIdx.x * 256 + threadIdx.x;
    if (i < B_ * H_) num[i] = 0.f;
    if (i < B_)      den[i] = 0.f;
}

// K1: aqb[b][h] = ba[h] + sum_k [ht;ct][b,k] * W_aq[k,h]   (32 x 512 x 256)
__global__ __launch_bounds__(256) void aq_kernel(
    const float* __restrict__ ht, const float* __restrict__ ct,
    const float* __restrict__ Waq, const float* __restrict__ ba,
    float* __restrict__ aqb)
{
    const int b = blockIdx.x, h = threadIdx.x;
    float s = ba[h];
#pragma unroll 4
    for (int k = 0; k < H_; ++k)
        s = fmaf(ht[b * H_ + k], Waq[k * H_ + h], s);            // scalar ht, coalesced W
#pragma unroll 4
    for (int k = 0; k < H_; ++k)
        s = fmaf(ct[b * H_ + k], Waq[(H_ + k) * H_ + h], s);
    aqb[b * H_ + h] = s;
}

// K2: Wt[n][k] = bf16(W_ah[k][n])  — 128KB, stays L2-resident.
__global__ __launch_bounds__(256) void wt_kernel(
    const float* __restrict__ Wah, unsigned short* __restrict__ Wt)
{
    const int n = blockIdx.x, k = threadIdx.x;
    Wt[n * H_ + k] = f32_to_bf16(Wah[k * H_ + n]);
}

// K3: fully fused:  ah-GEMM (bf16 WMMA) + tanh + v-dot + exp + weighted hi-sum.
// grid (32, 32): block = (batch b, 256-row T chunk); 8 waves x 2 m-tiles x 16 rows.
// hi crosses HBM exactly once; the epilogue re-read of each row hits WGP$/L2.
__global__ __launch_bounds__(256) void wmma_attn_kernel(
    const float* __restrict__ hi, const unsigned short* __restrict__ Wt,
    const float* __restrict__ aqb, const float* __restrict__ v,
    float* __restrict__ num, float* __restrict__ den)
{
    extern __shared__ __align__(16) char smem[];
    unsigned short* Wlds   = (unsigned short*)smem;        // [256][WPAD] bf16
    float*          numLDS = (float*)(smem + NUM_OFF);     // [256]
    float*          denLDS = (float*)(smem + DEN_OFF);     // [1]

    const int tid  = threadIdx.x;
    const int lane = tid & 31;
    const int wave = tid >> 5;
    const int b    = blockIdx.x;
    const int t0   = blockIdx.y * 256;

    // Stage Wt (row-major [n][k], k contiguous) into padded LDS via CDNA5 async DMA.
    for (int c = tid; c < 8192; c += 256) {                // 8192 chunks of 8 bf16 (16B)
        const int n  = c >> 5;
        const int k8 = (c & 31) << 3;
#if defined(HAVE_ASYNC_LDS)
        __builtin_amdgcn_global_load_async_to_lds_b128(
            (g_v4i_ptr)(Wt + (size_t)c * 8),
            (l_v4i_ptr)&Wlds[n * WPAD + k8],
            0, 0);
#else
        uint4 d = ((const uint4*)Wt)[c];
        *(uint4*)&Wlds[n * WPAD + k8] = d;
#endif
    }
    numLDS[tid] = 0.f;
    if (tid == 0) denLDS[0] = 0.f;
    wait_async_then_barrier();

    const int r = lane & 15;        // row within m-tile (A) / col within n-tile (B,C)
    const int g = lane >> 4;        // K-half selector per ISA 16-bit A/B layouts
    const float* aqb_b = aqb + b * H_;

    float acc8[8] = {0, 0, 0, 0, 0, 0, 0, 0};   // num partials, cols lane*8 .. lane*8+7
    float dsum = 0.f;                           // sum of ex over this wave's 32 rows

#pragma unroll 1
    for (int mt = 0; mt < 2; ++mt) {
        const int rowbase = t0 + wave * 32 + mt * 16;
        const float* hrow = hi + ((size_t)b * T_ + rowbase + r) * H_;
        __builtin_prefetch(hrow + 16 * H_, 0, 1);          // global_prefetch_b8

        // ---- Build bf16 A operands (16x32 per k-step), ISA lane layout:
        // lane<16: row r, K = 32kk + 8g + {0..7} and +16; lane>=16 same rows, other K half.
        v16bf A[8];
#pragma unroll
        for (int kk = 0; kk < 8; ++kk) {
            const int k0 = kk * 32 + g * 8;
            float4 f0 = *(const float4*)(hrow + k0);
            float4 f1 = *(const float4*)(hrow + k0 + 4);
            float4 f2 = *(const float4*)(hrow + k0 + 16);
            float4 f3 = *(const float4*)(hrow + k0 + 20);
            U32x8 u;
            u.u[0] = pack2bf(f0.x, f0.y); u.u[1] = pack2bf(f0.z, f0.w);
            u.u[2] = pack2bf(f1.x, f1.y); u.u[3] = pack2bf(f1.z, f1.w);
            u.u[4] = pack2bf(f2.x, f2.y); u.u[5] = pack2bf(f2.z, f2.w);
            u.u[6] = pack2bf(f3.x, f3.y); u.u[7] = pack2bf(f3.z, f3.w);
            A[kk] = __builtin_bit_cast(v16bf, u);
        }

        float p[8] = {0, 0, 0, 0, 0, 0, 0, 0};
        // unroll 1: keeps B operands flowing from LDS (full unroll spilled in round 1)
#pragma unroll 1
        for (int nt = 0; nt < 16; ++nt) {
            const int n     = nt * 16 + r;
            const float aqn = aqb_b[n];                     // includes ba
            const float vn  = v[n];
            v8f acc = {0.f, 0.f, 0.f, 0.f, 0.f, 0.f, 0.f, 0.f};
            const unsigned short* wp = &Wlds[n * WPAD + g * 8];
#pragma unroll
            for (int kk = 0; kk < 8; ++kk) {
                // B operand 32x16: lane holds col n, K = 32kk + 8g + {0..7} and +16 (2x ds_load_b128)
                U32x8 u;
                *(uint4*)&u.u[0] = *(const uint4*)(wp + kk * 32);
                *(uint4*)&u.u[4] = *(const uint4*)(wp + kk * 32 + 16);
                v16bf Bv = __builtin_bit_cast(v16bf, u);
                acc = __builtin_amdgcn_wmma_f32_16x16x32_bf16(
                    false, A[kk], false, Bv, (short)0, acc, false, false);
            }
            // C layout: VGPR i -> M = i + 8g, N = n. Fuse tanh + v-dot.
#pragma unroll
            for (int i = 0; i < 8; ++i) {
                float val = fast_tanh(acc[i] + aqn);
                p[i] = fmaf(vn, val, p[i]);
            }
        }
        // Row score = sum over all 256 n: reduce the 16 lanes of each half.
#pragma unroll
        for (int i = 0; i < 8; ++i) {
#pragma unroll
            for (int off = 1; off < 16; off <<= 1)
                p[i] += __shfl_xor(p[i], off, 32);
        }
        // Unstabilized exp (faithful to reference). p[i] = score of row i + 8g.
        float ex[8], exo[8];
#pragma unroll
        for (int i = 0; i < 8; ++i) {
            ex[i]  = __builtin_amdgcn_exp2f(p[i] * LOG2E_F);
            exo[i] = __shfl_xor(ex[i], 16, 32);             // other half's rows
        }
        // ---- Fused weighted sum: acc8[q] += ex[row] * hi[row][lane*8+q].
        // Per-row re-read is fully coalesced (wave covers one 1KB row) and L2-hot.
        const float* hrow2 = hi + ((size_t)b * T_ + rowbase) * H_ + lane * 8;
#pragma unroll
        for (int j = 0; j < 16; ++j) {
            const int half = j >> 3, i = j & 7;
            const float er = (half == g) ? ex[i] : exo[i];  // per-lane v_cndmask
            float4 h0 = *(const float4*)(hrow2 + (size_t)j * H_);
            float4 h1 = *(const float4*)(hrow2 + (size_t)j * H_ + 4);
            acc8[0] = fmaf(er, h0.x, acc8[0]); acc8[1] = fmaf(er, h0.y, acc8[1]);
            acc8[2] = fmaf(er, h0.z, acc8[2]); acc8[3] = fmaf(er, h0.w, acc8[3]);
            acc8[4] = fmaf(er, h1.x, acc8[4]); acc8[5] = fmaf(er, h1.y, acc8[5]);
            acc8[6] = fmaf(er, h1.z, acc8[6]); acc8[7] = fmaf(er, h1.w, acc8[7]);
            dsum += er;                                     // every lane: same total
        }
    }

    // Cross-wave reduction in LDS (ds_add_f32), then one global atomic per column.
    const int c0 = lane * 8;
#pragma unroll
    for (int q = 0; q < 8; ++q)
        atomicAdd(&numLDS[c0 + q], acc8[q]);                // lanes hit distinct cols
    if (lane == 0)
        atomicAdd(&denLDS[0], dsum);
    __syncthreads();

    atomicAdd(num + b * H_ + tid, numLDS[tid]);
    if (tid == 0) atomicAdd(den + b, denLDS[0]);
}

// K4: e[b,h] = num[b,h] / den[b]
__global__ __launch_bounds__(256) void fin_kernel(
    const float* __restrict__ num, const float* __restrict__ den,
    float* __restrict__ out)
{
    const int b = blockIdx.x, h = threadIdx.x;
    out[b * H_ + h] = num[b * H_ + h] / den[b];
}

// ---------------------------------------------------------------------------
extern "C" void kernel_launch(void* const* d_in, const int* in_sizes, int n_in,
                              void* d_out, int out_size, void* d_ws, size_t ws_size,
                              hipStream_t stream) {
    const float* ht  = (const float*)d_in[0];
    const float* ct  = (const float*)d_in[1];
    const float* hi  = (const float*)d_in[2];
    const float* Waq = (const float*)d_in[3];
    const float* Wah = (const float*)d_in[4];
    const float* ba  = (const float*)d_in[5];
    const float* v   = (const float*)d_in[6];
    float* out = (float*)d_out;

    char* ws = (char*)d_ws;
    float*          aqb = (float*)ws;                              // 32 KB
    unsigned short* Wt  = (unsigned short*)(ws + (32u << 10));     // 128 KB
    float*          num = (float*)(ws + (160u << 10));             // 32 KB
    float*          den = num + B_ * H_;                           // 128 B

    init_kernel<<<32, 256, 0, stream>>>(num, den);
    aq_kernel<<<B_, 256, 0, stream>>>(ht, ct, Waq, ba, aqb);
    wt_kernel<<<H_, 256, 0, stream>>>(Wah, Wt);

    wmma_attn_kernel<<<dim3(B_, T_ / 256), 256, LDS_BYTES, stream>>>(
        hi, Wt, aqb, v, num, den);                                 // ~133 KB LDS / 320 KB WGP

    fin_kernel<<<B_, 256, 0, stream>>>(num, den, out);
}
